// GNN_37692632990116
// MI455X (gfx1250) — matrix operand order repaired and checked by reading, not compiled, not run on hardware
//
#include <hip/hip_runtime.h>
#include <hip/hip_bf16.h>

typedef __bf16 bf16;
typedef __attribute__((ext_vector_type(16))) __bf16 v16bf;
typedef __attribute__((ext_vector_type(8)))  __bf16 v8bf;
typedef __attribute__((ext_vector_type(8)))  float  v8f;

constexpr int Nn   = 50000;
constexpr int Ee   = 800000;
constexpr int Hd   = 128;
constexpr int Ll   = 4;
constexpr int Kk   = 5;
constexpr int Bb   = 64;

constexpr int WROW = 136;   // padded LDS row stride (bf16 elems): 272B -> 4-bank lane advance

__device__ __forceinline__ bf16 f2bf(float f) {
  unsigned u = __builtin_bit_cast(unsigned, f);
  unsigned r = (u + 0x7FFFu + ((u >> 16) & 1u)) >> 16;
  unsigned short s = (unsigned short)r;
  return __builtin_bit_cast(bf16, s);
}

// 8 consecutive K-elements of the A operand (concat [A0|A1]) with optional
// row scale (A0 only) and per-column BN+ReLU fusion.
__device__ __forceinline__ void load_run8(float* r,
    const float* __restrict__ A0, const float* __restrict__ A1,
    int arow, int gk, float rs,
    const float* __restrict__ bnscale, const float* __restrict__ bnshift) {
  if (gk < Hd) {
    const float4* p = (const float4*)(A0 + (size_t)arow * Hd + gk);
    float4 a = p[0], b = p[1];
    r[0]=a.x; r[1]=a.y; r[2]=a.z; r[3]=a.w; r[4]=b.x; r[5]=b.y; r[6]=b.z; r[7]=b.w;
    #pragma unroll
    for (int e = 0; e < 8; ++e) {
      float v = r[e] * rs;
      if (bnscale) { v = v * bnscale[gk + e] + bnshift[gk + e]; v = v > 0.f ? v : 0.f; }
      r[e] = v;
    }
  } else {
    const float4* p = (const float4*)(A1 + (size_t)arow * Hd + (gk - Hd));
    float4 a = p[0], b = p[1];
    r[0]=a.x; r[1]=a.y; r[2]=a.z; r[3]=a.w; r[4]=b.x; r[5]=b.y; r[6]=b.z; r[7]=b.w;
  }
}

// ---------------------------------------------------------------------------
// WMMA GEMM: C[n, 0:128] = [A0 | A1][n, :] @ W + bias.
// WT is the bf16 weight, pre-transposed to [Hd][KA] so both the LDS stage and
// the per-lane B-fragment reads are contiguous (ds_load_b128 pairs).
// Block: 256 threads = 8 waves; wave owns 16 rows x 128 cols = 8 WMMA accums.
// ---------------------------------------------------------------------------
__global__ __launch_bounds__(256) void gemm_bf16_k(
    const float* __restrict__ A0, const float* __restrict__ A1,
    const float* __restrict__ rowscale0,
    const float* __restrict__ bnscale, const float* __restrict__ bnshift,
    const bf16*  __restrict__ WT, const float* __restrict__ bias,
    float* __restrict__ C, int nrows, int KA)
{
  __shared__ __align__(16) bf16 s_wt[Hd * WROW];   // ~34 KB transposed tile
  const int tid  = threadIdx.x;
  const int lane = tid & 31;
  const int wv   = tid >> 5;                       // wave 0..7
  const int l16  = lane & 15;
  const int hi   = lane >> 4;                      // 0/1 half of wave
  const int rowBase = blockIdx.x * 128 + wv * 16;
  int arow = rowBase + l16;
  if (arow > nrows - 1) arow = nrows - 1;          // clamp loads; stores guarded
  const float rs = rowscale0 ? rowscale0[arow] : 1.0f;

  v8f acc[8];
  #pragma unroll
  for (int ct = 0; ct < 8; ++ct)
    #pragma unroll
    for (int j = 0; j < 8; ++j) acc[ct][j] = 0.0f;

  for (int kh = 0; kh < KA; kh += 128) {
    // stage 128 K of all 128 cols: contiguous uint4 reads, padded uint4 writes
    #pragma unroll
    for (int it = 0; it < 8; ++it) {
      const int idx = it * 256 + tid;              // 2048 uint4 total
      const int c = idx >> 4, kk4 = idx & 15;
      ((uint4*)s_wt)[c * (WROW >> 3) + kk4] =
          ((const uint4*)(WT + (size_t)c * KA + kh))[kk4];
    }
    __syncthreads();

    #pragma unroll
    for (int k0 = 0; k0 < 128; k0 += 32) {
      // A fragment (ISA 7.12.2): two 8-contiguous K runs per lane
      float r0[8], r1[8];
      const int gkb = kh + k0 + hi * 8;
      load_run8(r0, A0, A1, arow, gkb,      rs, bnscale, bnshift);
      load_run8(r1, A0, A1, arow, gkb + 16, rs, bnscale, bnshift);
      v16bf afrag;
      #pragma unroll
      for (int e = 0; e < 8; ++e) { afrag[e] = f2bf(r0[e]); afrag[e + 8] = f2bf(r1[e]); }

      #pragma unroll
      for (int ct = 0; ct < 8; ++ct) {
        const bf16* bp = s_wt + (ct * 16 + l16) * WROW + k0 + hi * 8;
        const v8bf lo = *(const v8bf*)bp;          // ds_load_b128
        const v8bf hi8 = *(const v8bf*)(bp + 16);  // ds_load_b128
        const v16bf bfrag = __builtin_shufflevector(lo, hi8,
            0,1,2,3,4,5,6,7,8,9,10,11,12,13,14,15);
        acc[ct] = __builtin_amdgcn_wmma_f32_16x16x32_bf16(
            false, afrag, false, bfrag, (short)0, acc[ct], false, false);
      }
    }
    __syncthreads();
  }

  // C/D layout: VGPR j -> M = j (lanes 0-15) or 8+j (lanes 16-31), N = lane&15
  #pragma unroll
  for (int ct = 0; ct < 8; ++ct) {
    const int col = ct * 16 + l16;
    const float bb = bias ? bias[col] : 0.0f;
    #pragma unroll
    for (int j = 0; j < 8; ++j) {
      const int row = rowBase + hi * 8 + j;
      if (row < nrows) C[row * Hd + col] = acc[ct][j] + bb;
    }
  }
}

// ---------------------- BN batch-stats (biased variance) --------------------
__global__ __launch_bounds__(256) void bn_partial_k(const float* __restrict__ X,
                                                    float* __restrict__ sums, int nrows) {
  __shared__ float sh[512];
  const int tid = threadIdx.x, c = tid & 127, half = tid >> 7;
  const int rowsPer = (nrows + (int)gridDim.x - 1) / (int)gridDim.x;
  const int r0 = blockIdx.x * rowsPer;
  int r1 = r0 + rowsPer; if (r1 > nrows) r1 = nrows;
  float s = 0.f, q = 0.f;
  for (int r = r0 + half; r < r1; r += 2) {
    const float v = X[r * Hd + c];
    s += v; q += v * v;
  }
  sh[tid] = s; sh[256 + tid] = q;
  __syncthreads();
  if (tid < 128) {
    atomicAdd(&sums[tid],       sh[tid] + sh[tid + 128]);
    atomicAdd(&sums[128 + tid], sh[256 + tid] + sh[384 + tid]);
  }
}

__global__ void bn_final_k(const float* __restrict__ sums, const float* __restrict__ g,
                           const float* __restrict__ be, float* __restrict__ scale,
                           float* __restrict__ shift, float inv_n) {
  const int c = threadIdx.x;
  const float m  = sums[c] * inv_n;
  const float v  = sums[128 + c] * inv_n - m * m;
  const float sc = g[c] * rsqrtf(v + 1e-5f);
  scale[c] = sc;
  shift[c] = be[c] - m * sc;
}

__global__ void bn_apply_relu_k(const float* __restrict__ X, const float* __restrict__ scale,
                                const float* __restrict__ shift, float* __restrict__ Y, int n) {
  const int i = blockIdx.x * 256 + threadIdx.x;
  if (i >= n) return;
  const int c = i & 127;
  float v = X[i] * scale[c] + shift[c];
  Y[i] = v > 0.f ? v : 0.f;
}

// ------------------------------ graph ops ----------------------------------
__global__ void deg_k(const int* __restrict__ dst, float* __restrict__ deg) {
  const int e = blockIdx.x * 256 + threadIdx.x;
  if (e < Ee) atomicAdd(&deg[dst[e]], 1.0f);
}
__global__ void invdeg_k(const float* __restrict__ deg, float* __restrict__ inv) {
  const int n = blockIdx.x * 256 + threadIdx.x;
  if (n < Nn) inv[n] = deg[n] > 0.f ? 1.0f / deg[n] : 0.0f;
}
__global__ __launch_bounds__(256) void scatter_k(const float* __restrict__ h,
    const int* __restrict__ src, const int* __restrict__ dst,
    float* __restrict__ agg) {
  const int gid = blockIdx.x * 256 + threadIdx.x;   // Ee*32 = 25.6M, fits int
  const int e = gid >> 5, q = gid & 31;
  if (e >= Ee) return;
  const float4 v = ((const float4*)(h + (size_t)src[e] * Hd))[q];
  float* d = agg + (size_t)dst[e] * Hd + q * 4;
  atomicAdd(d + 0, v.x); atomicAdd(d + 1, v.y);
  atomicAdd(d + 2, v.z); atomicAdd(d + 3, v.w);
}

// ----------------------------- misc elementwise ----------------------------
__global__ void init_h_k(const int* __restrict__ x, const float* __restrict__ emb,
                         float* __restrict__ h) {
  const int i = blockIdx.x * 256 + threadIdx.x;
  if (i >= Nn * Hd) return;
  const int n = i >> 7, c = i & 127;
  h[i] = emb[(size_t)x[n] * Hd + c];
}
__global__ void unatom_k(const float* __restrict__ mask, int* __restrict__ ua) {
  const int n = blockIdx.x * 256 + threadIdx.x;
  if (n < Nn) ua[n] = (mask[n] == 0.0f) ? 2 : 0;
}
__global__ void t_init_k(const int* __restrict__ ua, const float* __restrict__ semb,
                         float* __restrict__ t) {
  const int i = blockIdx.x * 256 + threadIdx.x;
  if (i >= Nn * Hd) return;
  const int n = i >> 7, c = i & 127;
  t[i] = semb[ua[n] * Hd + c] + semb[3 * Hd + c];
}
__global__ void t_update_k(const int* __restrict__ trace, const int* __restrict__ inloop,
                           const int* __restrict__ ua, const float* __restrict__ semb,
                           float* __restrict__ t, int step) {
  const int i = blockIdx.x * 256 + threadIdx.x;
  if (i >= Nn * Hd) return;
  const int n = i >> 7, c = i & 127;
  const int i1 = trace[n * Kk + step] + ua[n];
  const int i2 = inloop[n * Kk + step] + 3;
  t[i] = semb[i1 * Hd + c] + semb[i2 * Hd + c];
}
__global__ void states_k(const float* __restrict__ h, const int* __restrict__ uidx,
                         float* __restrict__ st, int step) {
  const int i = blockIdx.x * 256 + threadIdx.x;
  if (i >= Bb * Hd) return;
  const int b = i >> 7, c = i & 127;
  st[((size_t)b * Kk + step) * Hd + c] = h[(size_t)uidx[b] * Hd + c];
}

// ------------------------------- heads -------------------------------------
__global__ __launch_bounds__(256) void ar_head_k(const float* __restrict__ Y,
    const float* __restrict__ scale, const float* __restrict__ shift,
    const float* __restrict__ W2, const float* __restrict__ b2,
    const float* __restrict__ mask, float* __restrict__ atom_st, int step) {
  const int wv = threadIdx.x >> 5, lane = threadIdx.x & 31;
  const int n = blockIdx.x * 8 + wv;
  if (n >= Nn) return;
  float s0 = 0.f, s1 = 0.f;
  #pragma unroll
  for (int q = 0; q < 4; ++q) {
    const int c = lane * 4 + q;
    float v = Y[(size_t)n * Hd + c] * scale[c] + shift[c];
    v = v > 0.f ? v : 0.f;
    s0 += v * W2[c * 2 + 0];
    s1 += v * W2[c * 2 + 1];
  }
  for (int d = 16; d; d >>= 1) { s0 += __shfl_down(s0, d, 32); s1 += __shfl_down(s1, d, 32); }
  if (lane == 0) {
    const float m = mask[n];
    atom_st[((size_t)n * Kk + step) * 2 + 0] = (s0 + b2[0]) * m;
    atom_st[((size_t)n * Kk + step) * 2 + 1] = (s1 + b2[1]) * m;
  }
}

__global__ __launch_bounds__(256) void il_head_k(const float* __restrict__ Y,
    const float* __restrict__ scale, const float* __restrict__ shift,
    const float* __restrict__ W2, const float* __restrict__ b2,
    const int* __restrict__ ptime, float* __restrict__ out) {
  const int wv = threadIdx.x >> 5, lane = threadIdx.x & 31;
  const int b = blockIdx.x * 8 + wv;
  if (b >= Bb) return;
  const int row = b * Kk + ptime[b];
  float s0 = 0.f, s1 = 0.f;
  #pragma unroll
  for (int q = 0; q < 4; ++q) {
    const int c = lane * 4 + q;
    float v = Y[(size_t)row * Hd + c] * scale[c] + shift[c];
    v = v > 0.f ? v : 0.f;
    s0 += v * W2[c * 2 + 0];
    s1 += v * W2[c * 2 + 1];
  }
  for (int d = 16; d; d >>= 1) { s0 += __shfl_down(s0, d, 32); s1 += __shfl_down(s1, d, 32); }
  if (lane == 0) { out[b * 2 + 0] = s0 + b2[0]; out[b * 2 + 1] = s1 + b2[1]; }
}

__global__ void predict_trace_k(const float* __restrict__ atom_st,
                                const int* __restrict__ ptime, const int* __restrict__ batch,
                                float* __restrict__ out) {
  const int i = blockIdx.x * 256 + threadIdx.x;
  if (i >= Nn * 2) return;
  const int n = i >> 1, o = i & 1;
  const int pt = ptime[batch[n]];
  out[i] = atom_st[((size_t)n * Kk + pt) * 2 + o];
}

// ------------------- weight conversion (transposed bf16) -------------------
// src [KAr, Hd] (row-major)  ->  dst [Hd, KAr]  (dst[c*KAr + k] = src[k*Hd + c])
__global__ void conv_t_k(const float* __restrict__ src, bf16* __restrict__ dst, int KAr) {
  const int i = blockIdx.x * 256 + threadIdx.x;
  if (i >= KAr * Hd) return;
  const int c = i / KAr, k = i % KAr;
  dst[i] = f2bf(src[(size_t)k * Hd + c]);
}
// stacked per-layer [Wl; Wr] transposed -> dst[l][c][k], k<Hd from Wl else Wr
__global__ void sage_stack_t_k(const float* __restrict__ Wl, const float* __restrict__ Wr,
                               bf16* __restrict__ dst) {
  const int i = blockIdx.x * 256 + threadIdx.x;
  if (i >= Ll * Hd * 2 * Hd) return;
  const int l = i / (Hd * 2 * Hd);
  const int rem = i % (Hd * 2 * Hd);
  const int c = rem / (2 * Hd);
  const int k = rem % (2 * Hd);
  const float v = (k < Hd) ? Wl[(size_t)l * Hd * Hd + k * Hd + c]
                           : Wr[(size_t)l * Hd * Hd + (k - Hd) * Hd + c];
  dst[i] = f2bf(v);
}

// ===========================================================================
extern "C" void kernel_launch(void* const* d_in, const int* in_sizes, int n_in,
                              void* d_out, int out_size, void* d_ws, size_t ws_size,
                              hipStream_t stream) {
  (void)in_sizes; (void)n_in; (void)out_size; (void)ws_size;

  const int*   x      = (const int*)  d_in[0];
  const int*   esrc   = (const int*)  d_in[1];
  const int*   edst   = esrc + Ee;
  const int*   trace  = (const int*)  d_in[2];
  const int*   batch  = (const int*)  d_in[3];
  const int*   uidx   = (const int*)  d_in[4];
  const int*   inloop = (const int*)  d_in[5];
  const int*   ptime  = (const int*)  d_in[6];
  const float* amask  = (const float*)d_in[7];
  const float* emb    = (const float*)d_in[8];
  const float* semb   = (const float*)d_in[9];
  const float* mlpW1  = (const float*)d_in[10];
  const float* mlpb1  = (const float*)d_in[11];
  const float* mlpg   = (const float*)d_in[12];
  const float* mlpbe  = (const float*)d_in[13];
  const float* mlpW2  = (const float*)d_in[14];
  const float* mlpb2  = (const float*)d_in[15];
  const float* sageWl = (const float*)d_in[16];
  const float* sagebl = (const float*)d_in[17];
  const float* sageWr = (const float*)d_in[18];
  const float* bng    = (const float*)d_in[19];
  const float* bnb    = (const float*)d_in[20];
  const float* arW1   = (const float*)d_in[21];
  const float* arb1   = (const float*)d_in[22];
  const float* arg    = (const float*)d_in[23];
  const float* arbe   = (const float*)d_in[24];
  const float* arW2   = (const float*)d_in[25];
  const float* arb2   = (const float*)d_in[26];
  const float* ilW1   = (const float*)d_in[27];
  const float* ilb1   = (const float*)d_in[28];
  const float* ilg    = (const float*)d_in[29];
  const float* ilbe   = (const float*)d_in[30];
  const float* ilW2   = (const float*)d_in[31];
  const float* ilb2   = (const float*)d_in[32];
  float* out = (float*)d_out;

  // ---- workspace carve-out ----
  char* wsb = (char*)d_ws;
  size_t off = 0;
  auto alloc = [&](size_t bytes) -> void* {
    void* p = wsb + off;
    off = (off + bytes + 255) & ~(size_t)255;
    return p;
  };
  float* hbuf    = (float*)alloc((size_t)Nn * Hd * 4);
  float* tbuf    = (float*)alloc((size_t)Nn * Hd * 4);
  float* zbuf    = (float*)alloc((size_t)Nn * Hd * 4);
  float* agg     = (float*)alloc((size_t)Nn * Hd * 4);
  float* atom_st = (float*)alloc((size_t)Nn * Kk * 2 * 4);
  float* stbuf   = (float*)alloc((size_t)Bb * Kk * Hd * 4);
  float* stz     = (float*)alloc((size_t)Bb * Kk * Hd * 4);
  float* deg     = (float*)alloc((size_t)Nn * 4);
  float* invdeg  = (float*)alloc((size_t)Nn * 4);
  int*   ua      = (int*)  alloc((size_t)Nn * 4);
  float* bn_sums = (float*)alloc(2 * Hd * 4);
  float* bn_sc   = (float*)alloc(Hd * 4);
  float* bn_sh   = (float*)alloc(Hd * 4);
  bf16*  w_mlp1  = (bf16*) alloc((size_t)2 * Hd * Hd * 2);   // [Hd][2Hd] transposed
  bf16*  w_mlp2  = (bf16*) alloc((size_t)Hd * Hd * 2);       // [Hd][Hd]
  bf16*  w_ar1   = (bf16*) alloc((size_t)Hd * Hd * 2);
  bf16*  w_il1   = (bf16*) alloc((size_t)Hd * Hd * 2);
  bf16*  w_sage  = (bf16*) alloc((size_t)Ll * Hd * 2 * Hd * 2); // [l][Hd][2Hd]

  const int NH = Nn * Hd;
  auto cdiv = [](int a, int b) { return (a + b - 1) / b; };

  auto gemm = [&](const float* A0, const float* A1, const float* rs0,
                  const float* bsc, const float* bsh,
                  const bf16* WT, const float* bias, float* C, int nrows, int KA) {
    gemm_bf16_k<<<cdiv(nrows, 128), 256, 0, stream>>>(A0, A1, rs0, bsc, bsh, WT, bias, C, nrows, KA);
  };
  auto bnstats = [&](const float* X, int nrows, const float* g, const float* be) {
    hipMemsetAsync(bn_sums, 0, 2 * Hd * sizeof(float), stream);
    bn_partial_k<<<128, 256, 0, stream>>>(X, bn_sums, nrows);
    bn_final_k<<<1, Hd, 0, stream>>>(bn_sums, g, be, bn_sc, bn_sh, 1.0f / (float)nrows);
  };

  // ---- one-time (per launch) precompute ----
  init_h_k<<<cdiv(NH, 256), 256, 0, stream>>>(x, emb, hbuf);
  hipMemsetAsync(deg, 0, (size_t)Nn * 4, stream);
  deg_k<<<cdiv(Ee, 256), 256, 0, stream>>>(edst, deg);
  invdeg_k<<<cdiv(Nn, 256), 256, 0, stream>>>(deg, invdeg);
  unatom_k<<<cdiv(Nn, 256), 256, 0, stream>>>(amask, ua);
  t_init_k<<<cdiv(NH, 256), 256, 0, stream>>>(ua, semb, tbuf);
  conv_t_k<<<cdiv(2 * Hd * Hd, 256), 256, 0, stream>>>(mlpW1, w_mlp1, 2 * Hd);
  conv_t_k<<<cdiv(Hd * Hd, 256), 256, 0, stream>>>(mlpW2, w_mlp2, Hd);
  conv_t_k<<<cdiv(Hd * Hd, 256), 256, 0, stream>>>(arW1, w_ar1, Hd);
  conv_t_k<<<cdiv(Hd * Hd, 256), 256, 0, stream>>>(ilW1, w_il1, Hd);
  sage_stack_t_k<<<cdiv(Ll * 2 * Hd * Hd, 256), 256, 0, stream>>>(sageWl, sageWr, w_sage);

  // ---- K unrolled time steps ----
  for (int i = 0; i < Kk; ++i) {
    // z = [h|t] @ mlp_W1 + b1 ; bn ; h = relu(bn(z)) @ mlp_W2 + b2
    gemm(hbuf, tbuf, nullptr, nullptr, nullptr, w_mlp1, mlpb1, zbuf, Nn, 2 * Hd);
    bnstats(zbuf, Nn, mlpg, mlpbe);
    gemm(zbuf, nullptr, nullptr, bn_sc, bn_sh, w_mlp2, mlpb2, hbuf, Nn, Hd);

    // 4 SAGE layers
    for (int l = 0; l < Ll; ++l) {
      hipMemsetAsync(agg, 0, (size_t)NH * 4, stream);
      scatter_k<<<cdiv(Ee * 32, 256), 256, 0, stream>>>(hbuf, esrc, edst, agg);
      gemm(agg, hbuf, invdeg, nullptr, nullptr, w_sage + (size_t)l * 2 * Hd * Hd,
           sagebl + l * Hd, zbuf, Nn, 2 * Hd);
      bnstats(zbuf, Nn, bng + l * Hd, bnb + l * Hd);
      bn_apply_relu_k<<<cdiv(NH, 256), 256, 0, stream>>>(zbuf, bn_sc, bn_sh, hbuf, NH);
    }

    // atom head: a = relu(bn(h@ar_W1+b1)) @ ar_W2 + b2, masked
    gemm(hbuf, nullptr, nullptr, nullptr, nullptr, w_ar1, arb1, zbuf, Nn, Hd);
    bnstats(zbuf, Nn, arg, arbe);
    ar_head_k<<<cdiv(Nn, 8), 256, 0, stream>>>(zbuf, bn_sc, bn_sh, arW2, arb2, amask, atom_st, i);

    // state gather + next-step t
    states_k<<<cdiv(Bb * Hd, 256), 256, 0, stream>>>(hbuf, uidx, stbuf, i);
    t_update_k<<<cdiv(NH, 256), 256, 0, stream>>>(trace, inloop, ua, semb, tbuf, i);
  }

  // ---- outputs ----
  predict_trace_k<<<cdiv(Nn * 2, 256), 256, 0, stream>>>(atom_st, ptime, batch, out);

  gemm(stbuf, nullptr, nullptr, nullptr, nullptr, w_il1, ilb1, stz, Bb * Kk, Hd);
  bnstats(stz, Bb * Kk, ilg, ilbe);
  il_head_k<<<cdiv(Bb, 8), 256, 0, stream>>>(stz, bn_sc, bn_sh, ilW2, ilb2, ptime, out + (size_t)Nn * 2);
}
